// LossLayer_35777077576046
// MI455X (gfx1250) — matrix operand order repaired and checked by reading, not compiled, not run on hardware
//
#include <hip/hip_runtime.h>
#include <math.h>

// ---------------------------------------------------------------------------
// Problem constants (from reference)
// ---------------------------------------------------------------------------
#define SS      48
#define NPIX    2304        // SS*SS
#define NROWS   640         // B*R*C = 16*8*5
#define HMC     5
#define HMH     160
#define HMW     160
#define RRECT   8
#define EPS_INV 10.0f       // 1/EPS
#define TINYV   1e-8f
#define ITERS   50

// GEMM tiling: WG = 128 thr = 4 waves; WG tile 128 rows x 64 cols;
// wave tile 32 rows x 64 cols = 2 row-tiles x 4 col-tiles = 8 WMMA / k-step.
#define BK          32      // k-step (WMMA depth)
#define BN          64      // cols per workgroup
#define COL_TILES   4
#define ROW_TILES   2
#define WAVES_PER_WG 4
#define ROWS_PER_WAVE 32
#define ROWS_PER_WG 128
#define LDS_STRIDE  80      // bytes per staged K-column (64B data + 16B pad)

typedef __bf16 bf16;
typedef __attribute__((ext_vector_type(16))) __bf16   v16bf;
typedef __attribute__((ext_vector_type(8)))  float    v8f;
typedef __attribute__((ext_vector_type(4)))  unsigned u32x4;
typedef __attribute__((ext_vector_type(4)))  int      v4i;

union Frag { v16bf v; u32x4 q[2]; };

// ---------------------------------------------------------------------------
// CDNA5 async global->LDS copy (ASYNCcnt path), with graceful fallbacks.
// ---------------------------------------------------------------------------
#if __has_builtin(__builtin_amdgcn_global_load_async_to_lds_b128)
#define HAVE_ASYNC_LDS 1
typedef __attribute__((address_space(1))) v4i* gp_v4i;
typedef __attribute__((address_space(3))) v4i* lp_v4i;
#else
#define HAVE_ASYNC_LDS 0
#endif

__device__ __forceinline__ void async_cp16(const void* g, void* l) {
#if HAVE_ASYNC_LDS
  __builtin_amdgcn_global_load_async_to_lds_b128(
      (gp_v4i)(unsigned long long)g,
      (lp_v4i)(unsigned)(unsigned long long)l, 0, 0);
#else
  *(u32x4*)l = *(const u32x4*)g;   // sync fallback: load_b128 + ds_store_b128
#endif
}

__device__ __forceinline__ void wait_async0() {
#if HAVE_ASYNC_LDS
#if __has_builtin(__builtin_amdgcn_s_wait_asynccnt)
  __builtin_amdgcn_s_wait_asynccnt(0);
#else
  asm volatile("s_wait_asynccnt 0x0" ::: "memory");
#endif
#endif
}

// ---------------------------------------------------------------------------
// Build K = exp(-dist/eps) and M = K*dist in bf16 (symmetric matrices)
// ---------------------------------------------------------------------------
__global__ void build_KM(bf16* __restrict__ Kb, bf16* __restrict__ Mb) {
  size_t idx = (size_t)blockIdx.x * blockDim.x + threadIdx.x;
  if (idx >= (size_t)NPIX * NPIX) return;
  int i = (int)(idx / NPIX);
  int j = (int)(idx % NPIX);
  float dy = (float)(i / SS - j / SS);
  float dx = (float)(i % SS - j % SS);
  float d  = sqrtf(dy * dy + dx * dx);
  float k  = __expf(-d * EPS_INV);     // underflows to 0 for far pixels (band matrix)
  Kb[idx] = (bf16)k;
  Mb[idx] = (bf16)(k * d);
}

// ---------------------------------------------------------------------------
// Crop + normalize marginals a (gt) / b (pred); init u = 1/n (bf16)
// ---------------------------------------------------------------------------
__global__ void crop_norm(const float* __restrict__ hm_gt,
                          const float* __restrict__ hm_pred,
                          const int*   __restrict__ tops,
                          float* __restrict__ a, float* __restrict__ b,
                          bf16*  __restrict__ u_bf) {
  const int row = blockIdx.x;                  // 0..639  == ((bb*R + rr)*C + cc)
  const int t   = threadIdx.x;                 // 0..255
  const int bb  = row / (RRECT * HMC);
  const int rr  = (row / HMC) % RRECT;
  const int cc  = row % HMC;
  const int y0  = tops[(bb * RRECT + rr) * 2 + 0];
  const int x0  = tops[(bb * RRECT + rr) * 2 + 1];
  const size_t plane = ((size_t)bb * HMC + cc) * HMH * HMW;
  const float* gbase = hm_gt   + plane;
  const float* pbase = hm_pred + plane;

  float gv[9], pv[9], sg = 0.f, sp = 0.f;
#pragma unroll
  for (int k = 0; k < 9; ++k) {
    int i  = t + k * 256;                      // 9*256 = 2304 exactly
    int iy = i / SS, ix = i % SS;
    float g = gbase[(size_t)(y0 + iy) * HMW + (x0 + ix)];
    float p = pbase[(size_t)(y0 + iy) * HMW + (x0 + ix)];
    gv[k] = g; pv[k] = p; sg += g; sp += p;
  }

  __shared__ float rg[256], rp[256];
  rg[t] = sg; rp[t] = sp; __syncthreads();
  for (int s = 128; s > 0; s >>= 1) {
    if (t < s) { rg[t] += rg[t + s]; rp[t] += rp[t + s]; }
    __syncthreads();
  }
  const float invg = 1.0f / (rg[0] + TINYV);
  const float invp = 1.0f / (rp[0] + TINYV);

  float* arow = a + (size_t)row * NPIX;
  float* brow = b + (size_t)row * NPIX;
  bf16*  urow = u_bf + (size_t)row * NPIX;
  const bf16 uinit = (bf16)(1.0f / (float)NPIX);
#pragma unroll
  for (int k = 0; k < 9; ++k) {
    int i = t + k * 256;
    arow[i] = gv[k] * invg;
    brow[i] = pv[k] * invp;
    urow[i] = uinit;
  }
}

// ---------------------------------------------------------------------------
// Fused GEMM: T = A[640 x 2304](bf16) @ Kmat[2304 x 2304](bf16, symmetric)
//   mode 0: Out = marg / (T + tiny)   (bf16)         -- Sinkhorn update
//   mode 1: lossPartial[wg] = sum(T * Vin)           -- final cost readout
// B tile double-buffered in LDS, staged with global_load_async_to_lds_b128
// using K's symmetry (column reads -> contiguous row reads).
// ---------------------------------------------------------------------------
__global__ void __launch_bounds__(128)
sinkhorn_gemm(const bf16*  __restrict__ A,
              const bf16*  __restrict__ Kmat,
              const float* __restrict__ marg,
              const bf16*  __restrict__ Vin,
              bf16*        __restrict__ Out,
              float*       __restrict__ lossPartial,
              int mode) {
  __shared__ alignas(16) unsigned char ldsB[2][BN * LDS_STRIDE];  // 2 x 5120 B
  __shared__ float red[128];

  const int tid   = threadIdx.x;
  const int wave  = tid >> 5;
  const int lane  = tid & 31;
  const int lhalf = lane >> 4;           // 0: lanes 0-15, 1: lanes 16-31
  const int l16   = lane & 15;

  const int col0 = blockIdx.x * BN;
  const int row0 = blockIdx.y * ROWS_PER_WG + wave * ROWS_PER_WAVE;

  // A fragment rows for this lane (A-matrix: lane%16 selects M)
  const bf16* aptr0 = A + (size_t)(row0 + l16) * NPIX;
  const bf16* aptr1 = aptr0 + (size_t)16 * NPIX;

  // B staging: 64 cols x 64 bytes = 4KB; each of 128 threads moves 32 bytes
  const int c    = tid >> 1;             // 0..63 (column within tile)
  const int half = tid & 1;              // 0..1  (32-byte half of the k-chunk)
  const bf16* krow = Kmat + (size_t)(col0 + c) * NPIX;   // symmetry: col -> row

  v8f acc[ROW_TILES][COL_TILES] = {};

  // prologue: stage k-block 0 into buffer 0
  async_cp16((const char*)krow + half * 32,
             &ldsB[0][c * LDS_STRIDE + half * 32]);
  async_cp16((const char*)krow + half * 32 + 16,
             &ldsB[0][c * LDS_STRIDE + half * 32 + 16]);

  int buf = 0;
  for (int kb = 0; kb < NPIX; kb += BK) {
    wait_async0();        // this wave's chunks of current buffer have landed
    __syncthreads();      // => every wave's chunks have landed; prior reads done

    // stage next k-block into the other buffer (overlaps with WMMA below)
    if (kb + BK < NPIX) {
      const char* src = (const char*)krow + (size_t)(kb + BK) * 2 + half * 32;
      unsigned char* dst = &ldsB[buf ^ 1][c * LDS_STRIDE + half * 32];
      async_cp16(src, dst);
      async_cp16(src + 16, dst + 16);
    }

    // --- A fragments (16x32 bf16): lanes<16 take K kb+{0..7,16..23}, else +8
    Frag af0, af1;
    {
      const char* ab0 = (const char*)aptr0 + (size_t)kb * 2 + (lhalf ? 16 : 0);
      const char* ab1 = (const char*)aptr1 + (size_t)kb * 2 + (lhalf ? 16 : 0);
      af0.q[0] = *(const u32x4*)(ab0);
      af0.q[1] = *(const u32x4*)(ab0 + 32);
      af1.q[0] = *(const u32x4*)(ab1);
      af1.q[1] = *(const u32x4*)(ab1 + 32);
    }

    // --- B fragments into distinct registers first (clause the LDS loads) ---
    Frag bfr[COL_TILES];
#pragma unroll
    for (int t2 = 0; t2 < COL_TILES; ++t2) {
      const unsigned char* lb =
          &ldsB[buf][(t2 * 16 + l16) * LDS_STRIDE + (lhalf ? 16 : 0)];
      bfr[t2].q[0] = *(const u32x4*)(lb);
      bfr[t2].q[1] = *(const u32x4*)(lb + 32);
    }

    // --- 8 WMMAs: each B fragment reused by both row-tiles ---
#pragma unroll
    for (int t2 = 0; t2 < COL_TILES; ++t2) {
      acc[0][t2] = __builtin_amdgcn_wmma_f32_16x16x32_bf16(
          false, af0.v, false, bfr[t2].v, (short)0, acc[0][t2], false, false);
      acc[1][t2] = __builtin_amdgcn_wmma_f32_16x16x32_bf16(
          false, af1.v, false, bfr[t2].v, (short)0, acc[1][t2], false, false);
    }
    buf ^= 1;
  }

  // --- epilogue: C/D layout => VGPR r holds M = r + 8*lhalf, N = lane%16 ---
  if (mode == 0) {
#pragma unroll
    for (int rt = 0; rt < ROW_TILES; ++rt) {
#pragma unroll
      for (int t2 = 0; t2 < COL_TILES; ++t2) {
        const int colb = col0 + t2 * 16 + l16;
#pragma unroll
        for (int r = 0; r < 8; ++r) {
          const int row = row0 + rt * 16 + r + lhalf * 8;
          const size_t off = (size_t)row * NPIX + colb;
          Out[off] = (bf16)(marg[off] / (acc[rt][t2][r] + TINYV));
        }
      }
    }
  } else {
    float s = 0.f;
#pragma unroll
    for (int rt = 0; rt < ROW_TILES; ++rt) {
#pragma unroll
      for (int t2 = 0; t2 < COL_TILES; ++t2) {
        const int colb = col0 + t2 * 16 + l16;
#pragma unroll
        for (int r = 0; r < 8; ++r) {
          const int row = row0 + rt * 16 + r + lhalf * 8;
          const size_t off = (size_t)row * NPIX + colb;
          s += acc[rt][t2][r] * (float)Vin[off];
        }
      }
    }
    red[tid] = s; __syncthreads();
    for (int st = 64; st > 0; st >>= 1) {
      if (tid < st) red[tid] += red[tid + st];
      __syncthreads();
    }
    if (tid == 0) lossPartial[blockIdx.y * gridDim.x + blockIdx.x] = red[0];
  }
}

// Deterministic single-block reduction of per-WG loss partials.
__global__ void final_reduce(const float* __restrict__ partials, int count,
                             float* __restrict__ out) {
  __shared__ float red[256];
  float s = 0.f;
  for (int i = threadIdx.x; i < count; i += 256) s += partials[i];
  red[threadIdx.x] = s; __syncthreads();
  for (int st = 128; st > 0; st >>= 1) {
    if (threadIdx.x < st) red[threadIdx.x] += red[threadIdx.x + st];
    __syncthreads();
  }
  if (threadIdx.x == 0) out[0] = red[0];
}

// ---------------------------------------------------------------------------
extern "C" void kernel_launch(void* const* d_in, const int* in_sizes, int n_in,
                              void* d_out, int out_size, void* d_ws, size_t ws_size,
                              hipStream_t stream) {
  const float* hm_gt   = (const float*)d_in[0];
  const float* hm_pred = (const float*)d_in[1];
  const int*   tops    = (const int*)d_in[2];
  float* out = (float*)d_out;

  const size_t n2 = (size_t)NPIX * NPIX;
  char* ws = (char*)d_ws;
  size_t off = 0;
  bf16*  Kbf = (bf16*)(ws + off);  off += n2 * sizeof(bf16);
  bf16*  Mbf = (bf16*)(ws + off);  off += n2 * sizeof(bf16);
  float* a   = (float*)(ws + off); off += (size_t)NROWS * NPIX * sizeof(float);
  float* b   = (float*)(ws + off); off += (size_t)NROWS * NPIX * sizeof(float);
  bf16*  u   = (bf16*)(ws + off);  off += (size_t)NROWS * NPIX * sizeof(bf16);
  bf16*  v   = (bf16*)(ws + off);  off += (size_t)NROWS * NPIX * sizeof(bf16);
  float* lossPartial = (float*)(ws + off);

  // 1) Gibbs kernel + cost matrix
  {
    int blocks = (int)((n2 + 255) / 256);
    build_KM<<<blocks, 256, 0, stream>>>(Kbf, Mbf);
  }
  // 2) crops + marginals + u0
  crop_norm<<<NROWS, 256, 0, stream>>>(hm_gt, hm_pred, tops, a, b, u);

  // 3) Sinkhorn iterations: K symmetric => both updates use the same GEMM
  const dim3 grid(NPIX / BN, NROWS / ROWS_PER_WG);   // 36 x 5
  for (int it = 0; it < ITERS; ++it) {
    // v = b / (u @ K + tiny)
    sinkhorn_gemm<<<grid, 128, 0, stream>>>(u, Kbf, b, nullptr, v, nullptr, 0);
    // u = a / (v @ K^T + tiny) ; K^T == K
    sinkhorn_gemm<<<grid, 128, 0, stream>>>(v, Kbf, a, nullptr, u, nullptr, 0);
  }

  // 4) loss = sum((u @ M) * v)
  sinkhorn_gemm<<<grid, 128, 0, stream>>>(u, Mbf, nullptr, v, nullptr, lossPartial, 1);
  final_reduce<<<1, 256, 0, stream>>>(lossPartial, (int)(grid.x * grid.y), out);
}
